// EncoderBlock_51402168598888
// MI455X (gfx1250) — matrix operand order repaired and checked by reading, not compiled, not run on hardware
//
#include <hip/hip_runtime.h>
#include <hip/hip_bf16.h>
#include <math.h>

// Problem sizes (fixed by the reference)
#define BB 4
#define SS 2048
#define DD 1024
#define HH 4096
#define ROWS (BB * SS)   // 8192
#define EPS 1e-6f

#define KTILE    64      // K depth staged per LDS buffer (2 WMMA sub-steps)
#define LDSPITCH 72      // halves per LDS tile row (64 data + 8 pad = 144B)

typedef __attribute__((ext_vector_type(16))) _Float16 v16h;
typedef __attribute__((ext_vector_type(8)))  _Float16 v8h;
typedef __attribute__((ext_vector_type(8)))  float    v8f;

// ---------------------------------------------------------------------------
// CDNA5 async global->LDS copy (16B per lane), tracked by ASYNCcnt.
// ISA operand order: VDST = LDS byte address VGPR, VADDR = 64-bit global addr.
// ---------------------------------------------------------------------------
__device__ __forceinline__ void async_copy_b128(const _Float16* g, unsigned lds) {
    asm volatile("global_load_async_to_lds_b128 %0, %1, off"
                 :: "v"(lds), "v"(g)
                 : "memory");
}
__device__ __forceinline__ void wait_async0() {
    asm volatile("s_wait_asynccnt 0x0" ::: "memory");
}
__device__ __forceinline__ unsigned lds_addr(const void* p) {
    return (unsigned)(unsigned long long)p;   // generic LDS ptr: addr[31:0] = LDS offset
}

// ---------------------------------------------------------------------------
// Block reduction helper (256 threads)
// ---------------------------------------------------------------------------
__device__ __forceinline__ float block_reduce_256(float v) {
    __shared__ float red[256];
    int tid = threadIdx.x;
    red[tid] = v;
    __syncthreads();
    for (int s = 128; s > 0; s >>= 1) {
        if (tid < s) red[tid] += red[tid + s];
        __syncthreads();
    }
    float r = red[0];
    __syncthreads();
    return r;
}

// ---------------------------------------------------------------------------
// LN1 applied only to row s=0 of each batch (the only row attention uses).
// ---------------------------------------------------------------------------
__global__ void ln1_row0_kernel(const float* __restrict__ x,
                                const float* __restrict__ a,
                                const float* __restrict__ bgam,
                                float* __restrict__ h0) {
    int b   = blockIdx.x;
    int tid = threadIdx.x;
    const float* xr = x + (size_t)b * SS * DD;   // s = 0
    float v[4], s = 0.f;
#pragma unroll
    for (int i = 0; i < 4; ++i) { v[i] = xr[tid + i * 256]; s += v[i]; }
    float mean = block_reduce_256(s) * (1.0f / DD);
    float s2 = 0.f;
#pragma unroll
    for (int i = 0; i < 4; ++i) { float d = v[i] - mean; s2 += d * d; }
    float var = block_reduce_256(s2) * (1.0f / (DD - 1));  // Bessel
    float inv = 1.0f / (sqrtf(var) + EPS);
#pragma unroll
    for (int i = 0; i < 4; ++i) {
        int c = tid + i * 256;
        h0[b * DD + c] = a[c] * (v[i] - mean) * inv + bgam[c];
    }
}

// ---------------------------------------------------------------------------
// Tiny GEMV: out[b][j] = sum_k h[b][k] * W[k][j]   (W is DDxDD row-major)
// ---------------------------------------------------------------------------
__global__ void gemv_kernel(const float* __restrict__ h,
                            const float* __restrict__ W,
                            float* __restrict__ out) {
    __shared__ float hs[DD];
    int b = blockIdx.y;
    int j = blockIdx.x * 256 + threadIdx.x;
    for (int i = threadIdx.x; i < DD; i += 256) hs[i] = h[b * DD + i];
    __syncthreads();
    float acc = 0.f;
#pragma unroll 8
    for (int k = 0; k < DD; ++k) acc += hs[k] * W[(size_t)k * DD + j];
    out[b * DD + j] = acc;
}

// ---------------------------------------------------------------------------
// Fused: x1 = x + broadcast(ao[b]);  h2 = f16(LN2(x1)).
// ---------------------------------------------------------------------------
__global__ void add_ln2_kernel(const float* __restrict__ x,
                               const float* __restrict__ ao,
                               const float* __restrict__ a2,
                               const float* __restrict__ b2g,
                               float* __restrict__ x1,
                               _Float16* __restrict__ h2) {
    int row = blockIdx.x;
    int b   = row / SS;
    int tid = threadIdx.x;
    const float* xr  = x  + (size_t)row * DD;
    const float* aor = ao + (size_t)b   * DD;
    float v[4], s = 0.f;
#pragma unroll
    for (int i = 0; i < 4; ++i) {
        int c = tid + i * 256;
        v[i] = xr[c] + aor[c];
        s += v[i];
    }
    float mean = block_reduce_256(s) * (1.0f / DD);
    float s2 = 0.f;
#pragma unroll
    for (int i = 0; i < 4; ++i) { float d = v[i] - mean; s2 += d * d; }
    float var = block_reduce_256(s2) * (1.0f / (DD - 1));
    float inv = 1.0f / (sqrtf(var) + EPS);
    float*     x1r = x1 + (size_t)row * DD;
    _Float16*  h2r = h2 + (size_t)row * DD;
#pragma unroll
    for (int i = 0; i < 4; ++i) {
        int c = tid + i * 256;
        x1r[c] = v[i];
        h2r[c] = (_Float16)(a2[c] * (v[i] - mean) * inv + b2g[c]);
    }
}

// ---------------------------------------------------------------------------
// Transpose + convert f32 [K][N] -> f16 [N][K].
// ---------------------------------------------------------------------------
__global__ void tconv_kernel(const float* __restrict__ w,
                             _Float16* __restrict__ wt,
                             int K, int N) {
    __shared__ float tile[32][33];
    int n0 = blockIdx.x * 32, k0 = blockIdx.y * 32;
    int tx = threadIdx.x, ty = threadIdx.y;
#pragma unroll
    for (int i = 0; i < 32; i += 8)
        tile[ty + i][tx] = w[(size_t)(k0 + ty + i) * N + n0 + tx];
    __syncthreads();
#pragma unroll
    for (int i = 0; i < 32; i += 8)
        wt[(size_t)(n0 + ty + i) * K + k0 + tx] = (_Float16)tile[tx][ty + i];
}

// ---------------------------------------------------------------------------
// GEMM core: 128x128 block tile, 4 waves (64x64 each, 4x4 wmma tiles).
// K staged 64-deep per LDS buffer via global_load_async_to_lds_b128
// (ASYNCcnt-tracked, double buffered, one barrier per 32 WMMAs); fragments
// read as conflict-free ds_load_b128 in the CDNA5 16-bit wave32 layouts:
//   A frag (16x32):  lane m=l&15,h=l>>4 : halves [8h..8h+7], [16+8h..23+8h]
//   B frag (32x16):  lane n=l&15,hg=l>>4: halves [8hg..8hg+7], [16+8hg..23+8hg] (N-major src)
// ---------------------------------------------------------------------------
__device__ __forceinline__ void gemm_core_128(
        const _Float16* __restrict__ A,   // [M][K]
        const _Float16* __restrict__ Bt,  // [N][K]
        int K, v8f acc[4][4]) {
    __shared__ __align__(16) _Float16 smem[2][2][128][LDSPITCH];

    const int tid  = threadIdx.x;
    const int lane = tid & 31;
    const int wave = tid >> 5;
    const int wr0  = (wave >> 1) * 64;   // wave row origin in block tile
    const int wc0  = (wave & 1) * 64;    // wave col origin in block tile

    const _Float16* Ag = A  + (size_t)(blockIdx.x * 128) * K;
    const _Float16* Bg = Bt + (size_t)(blockIdx.y * 128) * K;

    const int m  = lane & 15, h  = lane >> 4;   // A frag coords
    const int n  = lane & 15, hg = lane >> 4;   // B frag coords

    const int nk = K / KTILE;

    // stage one 128x64 A tile + 128x64 B tile: 1024 16B chunks each,
    // 8 per thread per matrix (16 async copies / thread / stage).
    auto stage = [&](int buf, int kb) {
#pragma unroll
        for (int j = 0; j < 8; ++j) {
            int chunk = tid + 128 * j;       // 0..1023
            int r = chunk >> 3, c = chunk & 7;
            async_copy_b128(Ag + (size_t)r * K + kb + 8 * c,
                            lds_addr(&smem[buf][0][r][c * 8]));
            async_copy_b128(Bg + (size_t)r * K + kb + 8 * c,
                            lds_addr(&smem[buf][1][r][c * 8]));
        }
    };

    stage(0, 0);
    for (int i = 0; i < nk; ++i) {
        wait_async0();        // this wave's copies for tile i landed
        __syncthreads();      // all waves' copies visible; prior reads retired
        if (i + 1 < nk) stage((i + 1) & 1, (i + 1) * KTILE);

        const _Float16(*As)[LDSPITCH] = smem[i & 1][0];
        const _Float16(*Bs)[LDSPITCH] = smem[i & 1][1];

#pragma unroll
        for (int ks = 0; ks < 2; ++ks) {          // two 32-deep sub-steps
            const int kofs = ks * 32;
            v16h a[4], b[4];
#pragma unroll
            for (int mi = 0; mi < 4; ++mi) {
                const _Float16* pa = &As[wr0 + mi * 16 + m][kofs + 8 * h];
                v8h lo = *(const v8h*)pa;
                v8h hi = *(const v8h*)(pa + 16);
#pragma unroll
                for (int e = 0; e < 8; ++e) { a[mi][e] = lo[e]; a[mi][8 + e] = hi[e]; }
            }
#pragma unroll
            for (int ni = 0; ni < 4; ++ni) {
                const _Float16* pb = &Bs[wc0 + ni * 16 + n][kofs + 8 * hg];
                v8h lo = *(const v8h*)pb;
                v8h hi = *(const v8h*)(pb + 16);
#pragma unroll
                for (int e = 0; e < 8; ++e) { b[ni][e] = lo[e]; b[ni][8 + e] = hi[e]; }
            }
#pragma unroll
            for (int ni = 0; ni < 4; ++ni)
#pragma unroll
                for (int mi = 0; mi < 4; ++mi)
                    acc[mi][ni] = __builtin_amdgcn_wmma_f32_16x16x32_f16(
                        false, a[mi], false, b[ni], (short)0, acc[mi][ni],
                        false, false);
        }
        // no bottom barrier: top barrier of iter i+1 orders reads vs refill
    }
}

// ---------------------------------------------------------------------------
// GEMM1: g = f16(gelu(h2 @ w1 + b1))
// ---------------------------------------------------------------------------
__global__ __launch_bounds__(128) void gemm1_gelu_kernel(
        const _Float16* __restrict__ A, const _Float16* __restrict__ Bt,
        const float* __restrict__ bias, _Float16* __restrict__ C,
        int M, int N, int K) {
    v8f acc[4][4];
#pragma unroll
    for (int i = 0; i < 4; ++i)
#pragma unroll
        for (int j = 0; j < 4; ++j) acc[i][j] = (v8f)(0.0f);

    gemm_core_128(A, Bt, K, acc);

    int lane = threadIdx.x & 31, wave = threadIdx.x >> 5;
    int row0 = blockIdx.x * 128 + (wave >> 1) * 64;
    int col0 = blockIdx.y * 128 + (wave & 1) * 64;
    int nlo = lane & 15, hl = lane >> 4;
#pragma unroll
    for (int mi = 0; mi < 4; ++mi)
#pragma unroll
        for (int ni = 0; ni < 4; ++ni) {
            int col = col0 + ni * 16 + nlo;
            float bv = bias[col];
#pragma unroll
            for (int r = 0; r < 8; ++r) {
                int row = row0 + mi * 16 + r + 8 * hl;
                float v = acc[mi][ni][r] + bv;
                float g = 0.5f * v * (1.0f + erff(v * 0.70710678118654752f));
                C[(size_t)row * N + col] = (_Float16)g;
            }
        }
}

// ---------------------------------------------------------------------------
// GEMM2: out = x1 + (g @ w2 + b2)
// ---------------------------------------------------------------------------
__global__ __launch_bounds__(128) void gemm2_res_kernel(
        const _Float16* __restrict__ A, const _Float16* __restrict__ Bt,
        const float* __restrict__ bias, const float* __restrict__ x1,
        float* __restrict__ out, int M, int N, int K) {
    v8f acc[4][4];
#pragma unroll
    for (int i = 0; i < 4; ++i)
#pragma unroll
        for (int j = 0; j < 4; ++j) acc[i][j] = (v8f)(0.0f);

    gemm_core_128(A, Bt, K, acc);

    int lane = threadIdx.x & 31, wave = threadIdx.x >> 5;
    int row0 = blockIdx.x * 128 + (wave >> 1) * 64;
    int col0 = blockIdx.y * 128 + (wave & 1) * 64;
    int nlo = lane & 15, hl = lane >> 4;
#pragma unroll
    for (int mi = 0; mi < 4; ++mi)
#pragma unroll
        for (int ni = 0; ni < 4; ++ni) {
            int col = col0 + ni * 16 + nlo;
            float bv = bias[col];
#pragma unroll
            for (int r = 0; r < 8; ++r) {
                int row = row0 + mi * 16 + r + 8 * hl;
                size_t idx = (size_t)row * N + col;
                out[idx] = x1[idx] + acc[mi][ni][r] + bv;
            }
        }
}

// ---------------------------------------------------------------------------
// Launch
// ---------------------------------------------------------------------------
extern "C" void kernel_launch(void* const* d_in, const int* in_sizes, int n_in,
                              void* d_out, int out_size, void* d_ws, size_t ws_size,
                              hipStream_t stream) {
    const float* x     = (const float*)d_in[0];
    // d_in[1]=wq, d_in[2]=wk : dead (attention is exactly one-hot on k=0)
    const float* wv    = (const float*)d_in[3];
    const float* wo    = (const float*)d_in[4];
    const float* w1    = (const float*)d_in[5];
    const float* b1    = (const float*)d_in[6];
    const float* w2    = (const float*)d_in[7];
    const float* b2    = (const float*)d_in[8];
    const float* ln1a  = (const float*)d_in[9];
    const float* ln1b  = (const float*)d_in[10];
    const float* ln2a  = (const float*)d_in[11];
    const float* ln2b  = (const float*)d_in[12];
    float* out = (float*)d_out;

    // workspace layout
    char* ws = (char*)d_ws;
    float*    x1   = (float*)ws;                 ws += (size_t)ROWS * DD * 4;  // 32 MB
    _Float16* h2   = (_Float16*)ws;              ws += (size_t)ROWS * DD * 2;  // 16 MB
    _Float16* w1t  = (_Float16*)ws;              ws += (size_t)DD * HH * 2;    //  8 MB
    _Float16* w2t  = (_Float16*)ws;              ws += (size_t)HH * DD * 2;    //  8 MB
    _Float16* g    = (_Float16*)ws;              ws += (size_t)ROWS * HH * 2;  // 64 MB
    float*    h0   = (float*)ws;                 ws += (size_t)BB * DD * 4;
    float*    v0   = (float*)ws;                 ws += (size_t)BB * DD * 4;
    float*    ao   = (float*)ws;                 ws += (size_t)BB * DD * 4;

    // --- weight convert+transpose ---
    tconv_kernel<<<dim3(HH / 32, DD / 32), dim3(32, 8), 0, stream>>>(w1, w1t, DD, HH);
    tconv_kernel<<<dim3(DD / 32, HH / 32), dim3(32, 8), 0, stream>>>(w2, w2t, HH, DD);

    // --- attention path (collapses to row-0 LN + two GEMVs) ---
    ln1_row0_kernel<<<BB, 256, 0, stream>>>(x, ln1a, ln1b, h0);
    gemv_kernel<<<dim3(DD / 256, BB), 256, 0, stream>>>(h0, wv, v0);
    gemv_kernel<<<dim3(DD / 256, BB), 256, 0, stream>>>(v0, wo, ao);

    // --- residual add + LN2 (fused), f16 activations for WMMA ---
    add_ln2_kernel<<<ROWS, 256, 0, stream>>>(x, ao, ln2a, ln2b, x1, h2);

    // --- FFN GEMMs on WMMA with async-LDS double buffering ---
    gemm1_gelu_kernel<<<dim3(ROWS / 128, HH / 128), 128, 0, stream>>>(
        h2, w1t, b1, g, ROWS, HH, DD);
    gemm2_res_kernel<<<dim3(ROWS / 128, DD / 128), 128, 0, stream>>>(
        g, w2t, b2, x1, out, ROWS, DD, HH);
}
